// BiLSTM_CRF_1056561955504
// MI455X (gfx1250) — matrix operand order repaired and checked by reading, not compiled, not run on hardware
//
#include <hip/hip_runtime.h>
#include <hip/hip_bf16.h>
#include <math.h>

typedef __attribute__((ext_vector_type(16))) __bf16 v16bf;
typedef __attribute__((ext_vector_type(8)))  __bf16 v8bf;
typedef __attribute__((ext_vector_type(8)))  float  v8f;

#define START_TAG 30
#define STOP_TAG  31
#define NEG_BIG   (-10000.0f)

// ---------------------------------------------------------------------------
// Gather embedding rows for the sentence and convert to bf16: X[t,d]
// ---------------------------------------------------------------------------
__global__ void gather_embed_bf16(const int* __restrict__ sent,
                                  const float* __restrict__ emb,
                                  __bf16* __restrict__ X, int D) {
  const int t = blockIdx.x;
  const size_t row = (size_t)sent[t];
  for (int d = threadIdx.x; d < D; d += blockDim.x)
    X[(size_t)t * D + d] = (__bf16)emb[row * D + d];
}

// ---------------------------------------------------------------------------
// f32 -> bf16 bulk convert
// ---------------------------------------------------------------------------
__global__ void cvt_f32_bf16(const float* __restrict__ src,
                             __bf16* __restrict__ dst, size_t n) {
  for (size_t i = blockIdx.x * (size_t)blockDim.x + threadIdx.x; i < n;
       i += (size_t)gridDim.x * blockDim.x)
    dst[i] = (__bf16)src[i];
}

// ---------------------------------------------------------------------------
// out[i] = a[i] + b[i]  (combined LSTM biases)
// ---------------------------------------------------------------------------
__global__ void vec_add(const float* __restrict__ a, const float* __restrict__ b,
                        float* __restrict__ out, int n) {
  int i = blockIdx.x * blockDim.x + threadIdx.x;
  if (i < n) out[i] = a[i] + b[i];
}

// ---------------------------------------------------------------------------
// Tile load helpers following CDNA5 ISA 7.12.2 per-lane WMMA layouts.
//   A (16x32 bf16): lane<16 holds K={0..7,16..23} of row lane,
//                   lane>=16 holds K={8..15,24..31}  -> two 16B loads 16 apart
//   B (32x16 bf16): same K order as A, column lane&15
//                   (lane>=16 -> K base +8)           -> two 16B loads 16 apart
// ---------------------------------------------------------------------------
__device__ __forceinline__ v16bf load_tile(const __bf16* p, int second_off) {
  v8bf lo = *(const v8bf*)p;
  v8bf hi = *(const v8bf*)(p + second_off);
  v16bf r;
#pragma unroll
  for (int i = 0; i < 8; ++i) { r[i] = lo[i]; r[i + 8] = hi[i]; }
  return r;
}

// ---------------------------------------------------------------------------
// C[M,N] = A[M,K] * W[N,K]^T + bias[N]   (bf16 in, f32 out)
// One wave per 32x32 output block (2x2 register blocking of 16x16 WMMA
// tiles): each A/B tile load feeds two v_wmma ops -> half the L2 bytes/MAC.
// M,N multiples of 32; K multiple of 32.
// ---------------------------------------------------------------------------
__global__ void gemm_bf16_wmma2x2(const __bf16* __restrict__ A,
                                  const __bf16* __restrict__ W,
                                  const float* __restrict__ bias,
                                  float* __restrict__ C,
                                  int M, int N, int K) {
  const int wave = (blockIdx.x * blockDim.x + threadIdx.x) >> 5;
  const int lane = threadIdx.x & 31;
  const int bN = N >> 5;
  const int bM = M >> 5;
  if (wave >= bM * bN) return;
  const int bm = wave / bN;
  const int bn = wave % bN;

  const int half = lane >> 4;   // 0: lanes 0-15, 1: lanes 16-31
  const int l16  = lane & 15;

  const __bf16* Ap0 = A + (size_t)(bm * 32 + l16)      * K + half * 8;
  const __bf16* Ap1 = A + (size_t)(bm * 32 + 16 + l16) * K + half * 8;
  const __bf16* Wp0 = W + (size_t)(bn * 32 + l16)      * K + half * 8;
  const __bf16* Wp1 = W + (size_t)(bn * 32 + 16 + l16) * K + half * 8;

  v8f acc00 = {}, acc01 = {}, acc10 = {}, acc11 = {};
  for (int k = 0; k < K; k += 32) {
    const v16bf a0 = load_tile(Ap0 + k, 16);
    const v16bf a1 = load_tile(Ap1 + k, 16);
    const v16bf b0 = load_tile(Wp0 + k, 16);
    const v16bf b1 = load_tile(Wp1 + k, 16);
    acc00 = __builtin_amdgcn_wmma_f32_16x16x32_bf16(false, a0, false, b0,
                                                    (short)0, acc00, false, false);
    acc01 = __builtin_amdgcn_wmma_f32_16x16x32_bf16(false, a0, false, b1,
                                                    (short)0, acc01, false, false);
    acc10 = __builtin_amdgcn_wmma_f32_16x16x32_bf16(false, a1, false, b0,
                                                    (short)0, acc10, false, false);
    acc11 = __builtin_amdgcn_wmma_f32_16x16x32_bf16(false, a1, false, b1,
                                                    (short)0, acc11, false, false);
  }

  // D layout: acc[r] -> row = rowbase + half*8 + r, col = colbase + l16
  const int c0 = bn * 32 + l16;
  const int c1 = c0 + 16;
  const float bv0 = bias ? bias[c0] : 0.0f;
  const float bv1 = bias ? bias[c1] : 0.0f;
#pragma unroll
  for (int r = 0; r < 8; ++r) {
    const int r0 = bm * 32 + half * 8 + r;
    const int r1 = r0 + 16;
    C[(size_t)r0 * N + c0] = acc00[r] + bv0;
    C[(size_t)r0 * N + c1] = acc01[r] + bv1;
    C[(size_t)r1 * N + c0] = acc10[r] + bv0;
    C[(size_t)r1 * N + c1] = acc11[r] + bv1;
  }
}

// ---------------------------------------------------------------------------
// LSTM recurrence, one workgroup (512 threads) per direction.
// G = precomputed  Wih@x + bih + bhh  for every timestep:  [L, 2048]
// Thread j owns hidden unit j: 4 dot-products (i,f,g,o rows of Whh) vs
// LDS-resident h, then the cell update.  dir 0 = forward, 1 = backward.
// Next timestep's gate row is prefetched (global_prefetch_b8) during the
// current dot products to hide the L2 round trip on the serial chain.
// ---------------------------------------------------------------------------
__global__ __launch_bounds__(512)
void lstm_recurrence(const float* __restrict__ Gf, const float* __restrict__ Gb,
                     const float* __restrict__ Whh_f, const float* __restrict__ Whh_b,
                     const float* __restrict__ h0, const float* __restrict__ c0,
                     float* __restrict__ Hcat, int L) {
  __shared__ float hsh[512];
  const int j = threadIdx.x;
  const int dir = blockIdx.x;
  const float* G   = dir ? Gb : Gf;
  const float* Whh = dir ? Whh_b : Whh_f;

  float c = c0[dir * 512 + j];
  hsh[j] = h0[dir * 512 + j];

  const float* Wi = Whh + (size_t)(j)        * 512;
  const float* Wf = Whh + (size_t)(512 + j)  * 512;
  const float* Wg = Whh + (size_t)(1024 + j) * 512;
  const float* Wo = Whh + (size_t)(1536 + j) * 512;
  __syncthreads();

  for (int s = 0; s < L; ++s) {
    const int t = dir ? (L - 1 - s) : s;
    const float* grow = G + (size_t)t * 2048;
    float ai = grow[j], af = grow[512 + j], ag = grow[1024 + j], ao = grow[1536 + j];

    if (s + 1 < L) {  // prefetch next step's gates while we compute
      const int tn = dir ? (L - 2 - s) : (s + 1);
      const float* gn = G + (size_t)tn * 2048;
      __builtin_prefetch(gn + j, 0, 1);
      __builtin_prefetch(gn + 512 + j, 0, 1);
      __builtin_prefetch(gn + 1024 + j, 0, 1);
      __builtin_prefetch(gn + 1536 + j, 0, 1);
    }

#pragma unroll 4
    for (int d = 0; d < 512; d += 4) {
      const float4 h4 = *(const float4*)&hsh[d];
      const float4 wi = *(const float4*)&Wi[d];
      const float4 wf = *(const float4*)&Wf[d];
      const float4 wg = *(const float4*)&Wg[d];
      const float4 wo = *(const float4*)&Wo[d];
      ai += h4.x * wi.x + h4.y * wi.y + h4.z * wi.z + h4.w * wi.w;
      af += h4.x * wf.x + h4.y * wf.y + h4.z * wf.z + h4.w * wf.w;
      ag += h4.x * wg.x + h4.y * wg.y + h4.z * wg.z + h4.w * wg.w;
      ao += h4.x * wo.x + h4.y * wo.y + h4.z * wo.z + h4.w * wo.w;
    }
    const float ig = 1.0f / (1.0f + __expf(-ai));
    const float fg = 1.0f / (1.0f + __expf(-af));
    const float gg = tanhf(ag);
    const float og = 1.0f / (1.0f + __expf(-ao));
    c = fg * c + ig * gg;
    const float h = og * tanhf(c);
    __syncthreads();          // all reads of previous h complete
    hsh[j] = h;
    Hcat[(size_t)t * 1024 + dir * 512 + j] = h;
    __syncthreads();          // new h visible
  }
}

// ---------------------------------------------------------------------------
// Viterbi over 32 tags with a single wave32: lane = destination tag.
// Writes score to out[0], path (as floats) to out[1..L].
// ---------------------------------------------------------------------------
__global__ __launch_bounds__(32)
void viterbi32(const float* __restrict__ feats, const float* __restrict__ trans,
               int* __restrict__ bp, float* __restrict__ out, int L) {
  __shared__ float tr[32 * 32];
  __shared__ float fv[32];
  __shared__ float term[32];
  const int to = threadIdx.x;
  for (int i = to; i < 32 * 32; i += 32) tr[i] = trans[i];
  fv[to] = (to == START_TAG) ? 0.0f : NEG_BIG;
  __syncthreads();

  for (int t = 0; t < L; ++t) {
    float best = -3.4e38f;
    int bj = 0;
#pragma unroll
    for (int from = 0; from < 32; ++from) {
      const float s = fv[from] + tr[from * 32 + to];
      if (s > best) { best = s; bj = from; }
    }
    bp[t * 32 + to] = bj;
    const float nv = best + feats[(size_t)t * 32 + to];
    __syncthreads();
    fv[to] = nv;
    __syncthreads();
  }

  term[to] = fv[to] + tr[to * 32 + STOP_TAG];
  __syncthreads();
  if (to == 0) {
    int best = 0; float bs = term[0];
    for (int i = 1; i < 32; ++i)
      if (term[i] > bs) { bs = term[i]; best = i; }
    out[0] = bs;
    int tag = best;
    out[1 + (L - 1)] = (float)best;
    for (int t = L - 1; t >= 1; --t) {
      tag = bp[t * 32 + tag];
      out[1 + (t - 1)] = (float)tag;
    }
  }
}

// ---------------------------------------------------------------------------
extern "C" void kernel_launch(void* const* d_in, const int* in_sizes, int n_in,
                              void* d_out, int out_size, void* d_ws, size_t ws_size,
                              hipStream_t stream) {
  (void)in_sizes; (void)n_in; (void)out_size; (void)ws_size;

  const int*   sentence = (const int*)  d_in[0];
  const float* emb      = (const float*)d_in[1];
  const float* Wih_f    = (const float*)d_in[2];
  const float* Whh_f    = (const float*)d_in[3];
  const float* bih_f    = (const float*)d_in[4];
  const float* bhh_f    = (const float*)d_in[5];
  const float* Wih_b    = (const float*)d_in[6];
  const float* Whh_b    = (const float*)d_in[7];
  const float* bih_b    = (const float*)d_in[8];
  const float* bhh_b    = (const float*)d_in[9];
  const float* W_out    = (const float*)d_in[10];
  const float* b_out    = (const float*)d_in[11];
  const float* trans    = (const float*)d_in[12];
  const float* h0       = (const float*)d_in[13];
  const float* c0       = (const float*)d_in[14];

  const int L = 2048, D = 512, H = 1024, G4 = 2048, T = 32;

  char* ws = (char*)d_ws;
  size_t off = 0;
  auto carve = [&](size_t bytes) -> void* {
    void* p = ws + off;
    off += (bytes + 255) & ~(size_t)255;
    return p;
  };
  __bf16* Xbf    = (__bf16*)carve((size_t)L * D * 2);       // gathered input, bf16
  __bf16* Wfbf   = (__bf16*)carve((size_t)G4 * D * 2);      // Wih_f bf16
  __bf16* Wbbf   = (__bf16*)carve((size_t)G4 * D * 2);      // Wih_b bf16
  __bf16* Wobf   = (__bf16*)carve((size_t)T * H * 2);       // W_out bf16
  __bf16* Hbf    = (__bf16*)carve((size_t)L * H * 2);       // hidden states bf16
  float*  Gf     = (float*) carve((size_t)L * G4 * 4);      // fwd input gates
  float*  Gb     = (float*) carve((size_t)L * G4 * 4);      // bwd input gates
  float*  Hcat   = (float*) carve((size_t)L * H * 4);       // [hf | hb]
  float*  feats  = (float*) carve((size_t)L * T * 4);       // emission scores
  int*    bp     = (int*)   carve((size_t)L * T * 4);       // viterbi backptrs
  float*  biasf  = (float*) carve((size_t)G4 * 4);          // bih_f + bhh_f
  float*  biasb  = (float*) carve((size_t)G4 * 4);          // bih_b + bhh_b

  // 1. gather + convert inputs / weights
  gather_embed_bf16<<<L, 256, 0, stream>>>(sentence, emb, Xbf, D);
  {
    size_t n = (size_t)G4 * D;
    int blocks = (int)((n + 255) / 256);
    cvt_f32_bf16<<<blocks, 256, 0, stream>>>(Wih_f, Wfbf, n);
    cvt_f32_bf16<<<blocks, 256, 0, stream>>>(Wih_b, Wbbf, n);
  }
  cvt_f32_bf16<<<(T * H + 255) / 256, 256, 0, stream>>>(W_out, Wobf, (size_t)T * H);
  vec_add<<<(G4 + 255) / 256, 256, 0, stream>>>(bih_f, bhh_f, biasf, G4);
  vec_add<<<(G4 + 255) / 256, 256, 0, stream>>>(bih_b, bhh_b, biasb, G4);

  // 2. batched input projections: G = X @ Wih^T + (bih + bhh)   [WMMA 2x2]
  {
    int waves  = (L / 32) * (G4 / 32);          // 4096 waves, 32x32 blocks
    int blocks = (waves * 32 + 255) / 256;
    gemm_bf16_wmma2x2<<<blocks, 256, 0, stream>>>(Xbf, Wfbf, biasf, Gf, L, G4, D);
    gemm_bf16_wmma2x2<<<blocks, 256, 0, stream>>>(Xbf, Wbbf, biasb, Gb, L, G4, D);
  }

  // 3. recurrences (fwd + bwd in parallel workgroups)
  lstm_recurrence<<<2, 512, 0, stream>>>(Gf, Gb, Whh_f, Whh_b, h0, c0, Hcat, L);

  // 4. emission projection: feats = Hcat @ W_out^T + b_out   [WMMA 2x2]
  cvt_f32_bf16<<<(int)(((size_t)L * H + 255) / 256), 256, 0, stream>>>(
      Hcat, Hbf, (size_t)L * H);
  {
    int waves  = (L / 32) * (T / 32);           // 64 waves
    int blocks = (waves * 32 + 255) / 256;
    gemm_bf16_wmma2x2<<<blocks, 256, 0, stream>>>(Hbf, Wobf, b_out, feats, L, T, H);
  }

  // 5. Viterbi decode (single wave32) + backtrack
  viterbi32<<<1, 32, 0, stream>>>(feats, trans, bp, (float*)d_out, L);
}
